// multihead_self_attention_73976516706485
// MI455X (gfx1250) — compile-verified, matrix operand order checked by reading, not run on hardware
//
#include <hip/hip_runtime.h>
#include <hip/hip_bf16.h>

// MI455X / gfx1250, wave32. Compute-bound MHSA: all matmuls through
// v_wmma_f32_16x16x32_f16 (f16 in, f32 accumulate). fp32->f16 once up front;
// V is stored transposed (B,H,DH,T) so every WMMA operand tile loads as
// contiguous 16B runs (global_load_b128) -- no strided gathers anywhere.

typedef __attribute__((ext_vector_type(16))) _Float16 v16h;
typedef __attribute__((ext_vector_type(4)))  _Float16 v4h;
typedef __attribute__((ext_vector_type(8)))  float    v8f;
typedef __attribute__((ext_vector_type(4)))  float    v4f;

#define B_  2
#define T_  2048
#define D_  1024
#define H_  16
#define DH_ 64

__device__ __forceinline__ v8f wmma16x16x32(v16h a, v16h b, v8f c) {
  // (neg_a, A, neg_b, B, c_mod, C, reuse_a, reuse_b)
  return __builtin_amdgcn_wmma_f32_16x16x32_f16(false, a, false, b, (short)0, c,
                                                false, false);
}

// A-matrix 16x32 (MxK), 16-bit, wave32 layout (ISA 7.12.2):
//   lanes 0-15: row M=lane, K = {0..7, 16..23}
//   lanes 16-31: row M=lane-16, K = {8..15, 24..31}
// Row-major f16 source; the two 8-half runs are contiguous 16B -> b128 loads.
// Also used for B-tiles given as B^T row-major (N-per-lane mirror layout).
__device__ __forceinline__ v16h load_lhs_16x32(const _Float16* __restrict__ p, int ld) {
  const int lane = threadIdx.x & 31;
  const int r  = lane & 15;
  const int kb = (lane >> 4) << 3;          // 0 or 8
  const _Float16* rp = p + (size_t)r * ld;
  v16h out;
#pragma unroll
  for (int e = 0; e < 8; ++e) {
    out[e]     = rp[kb + e];                // K = kb + e
    out[e + 8] = rp[16 + kb + e];           // K = 16 + kb + e
  }
  return out;
}

// ---------------------------------------------------------------------------
// Kernel 0: one-time fp32 -> f16 conversion (4 elems/thread, b128 in, b64 out)
// ---------------------------------------------------------------------------
__global__ __launch_bounds__(256)
void cvt_f32_to_f16_kernel(const float* __restrict__ src, _Float16* __restrict__ dst,
                           int n4) {
  const int i = blockIdx.x * blockDim.x + threadIdx.x;
  if (i < n4) {
    const v4f s = *(const v4f*)(src + (size_t)i * 4);
    v4h d;
    d[0] = (_Float16)s[0]; d[1] = (_Float16)s[1];
    d[2] = (_Float16)s[2]; d[3] = (_Float16)s[3];
    *(v4h*)(dst + (size_t)i * 4) = d;
  }
}

// ---------------------------------------------------------------------------
// Kernel 1: projection GEMM, C = scale * A(32xD) * W^T(Dx64-strip).
// One wave owns a 32(row) x 64(col) tile: two A tiles share every B tile.
// W^T tiles load with the A-pattern since B[k][n] = W[e0+n][k0+k].
// Stores f16 head-major (B,H,T,DH). Used for Q and K.
// ---------------------------------------------------------------------------
__global__ __launch_bounds__(32)
void proj_qkv_kernel(const _Float16* __restrict__ xh, const _Float16* __restrict__ w,
                     _Float16* __restrict__ out, float scale) {
  const int row0 = blockIdx.x * 32;     // row in flattened (B*T)
  const int h    = blockIdx.y;          // head = 64-col strip
  const int e0   = h * DH_;

  v8f acc[2][4] = {};
  for (int k0 = 0; k0 < D_; k0 += 32) {
    const v16h a0 = load_lhs_16x32(xh + (size_t)row0 * D_ + k0, D_);
    const v16h a1 = load_lhs_16x32(xh + (size_t)(row0 + 16) * D_ + k0, D_);
#pragma unroll
    for (int c = 0; c < 4; ++c) {
      const v16h b = load_lhs_16x32(w + (size_t)(e0 + c * 16) * D_ + k0, D_);
      acc[0][c] = wmma16x16x32(a0, b, acc[0][c]);
      acc[1][c] = wmma16x16x32(a1, b, acc[1][c]);
    }
  }

  const int b  = row0 / T_;
  const int t0 = row0 % T_;
  const int lane = threadIdx.x & 31;
  const int n  = lane & 15;
  const int mb = (lane >> 4) << 3;
#pragma unroll
  for (int rh = 0; rh < 2; ++rh) {
    _Float16* o = out + (((size_t)(b * H_ + h)) * T_ + t0 + rh * 16) * DH_;
#pragma unroll
    for (int c = 0; c < 4; ++c)
#pragma unroll
      for (int i = 0; i < 8; ++i)
        o[(size_t)(mb + i) * DH_ + c * 16 + n] = (_Float16)(acc[rh][c][i] * scale);
  }
}

// ---------------------------------------------------------------------------
// Kernel 1b: V projection storing V^T in (B,H,DH,T): vt[bh][dh][t].
// Same math; the store transposes (per lane, 8 consecutive t -> 16B runs).
// ---------------------------------------------------------------------------
__global__ __launch_bounds__(32)
void proj_vT_kernel(const _Float16* __restrict__ xh, const _Float16* __restrict__ w,
                    _Float16* __restrict__ vt) {
  const int row0 = blockIdx.x * 32;
  const int h    = blockIdx.y;
  const int e0   = h * DH_;

  v8f acc[2][4] = {};
  for (int k0 = 0; k0 < D_; k0 += 32) {
    const v16h a0 = load_lhs_16x32(xh + (size_t)row0 * D_ + k0, D_);
    const v16h a1 = load_lhs_16x32(xh + (size_t)(row0 + 16) * D_ + k0, D_);
#pragma unroll
    for (int c = 0; c < 4; ++c) {
      const v16h b = load_lhs_16x32(w + (size_t)(e0 + c * 16) * D_ + k0, D_);
      acc[0][c] = wmma16x16x32(a0, b, acc[0][c]);
      acc[1][c] = wmma16x16x32(a1, b, acc[1][c]);
    }
  }

  const int b  = row0 / T_;
  const int t0 = row0 % T_;
  const int lane = threadIdx.x & 31;
  const int n  = lane & 15;
  const int mb = (lane >> 4) << 3;
  _Float16* o = vt + ((size_t)(b * H_ + h)) * DH_ * T_;   // [dh][t]
#pragma unroll
  for (int rh = 0; rh < 2; ++rh)
#pragma unroll
    for (int c = 0; c < 4; ++c)
#pragma unroll
      for (int i = 0; i < 8; ++i)
        o[(size_t)(c * 16 + n) * T_ + t0 + rh * 16 + mb + i] = (_Float16)acc[rh][c][i];
}

// ---------------------------------------------------------------------------
// Kernel 2: causal flash attention per (b,h). One wave owns a 32-query tile
// (two 16-row halves sharing every K and V tile -> 2x K/V reuse), streams
// 32-key blocks: S = Q K^T (WMMA), online softmax (shfl_xor row reductions
// inside 16-lane groups), P staged C-layout -> A-layout via LDS (ds ops,
// same-wave in-order), O += P V^T-tiles (WMMA). Q pre-scaled by 1/sqrt(DH).
// ---------------------------------------------------------------------------
__global__ __launch_bounds__(32)
void flash_attn_kernel(const _Float16* __restrict__ qh,
                       const _Float16* __restrict__ kh,
                       const _Float16* __restrict__ vt,
                       _Float16* __restrict__ oh) {
  __shared__ _Float16 pbuf[32 * 40];    // 32 x 32 P tile, rows padded to 40 halves

  const int t0 = blockIdx.x * 32;       // query rows t0..t0+31
  const int bh = blockIdx.y;            // b*H + h
  const _Float16* Q  = qh + (size_t)bh * T_ * DH_;
  const _Float16* K  = kh + (size_t)bh * T_ * DH_;
  const _Float16* VT = vt + (size_t)bh * DH_ * T_;   // [dh][t]

  const int lane = threadIdx.x & 31;
  const int n  = lane & 15;
  const int mb = (lane >> 4) << 3;

  // Q tile (32 x 64) as four A operands, resident for the whole loop
  v16h aq[2][2];
#pragma unroll
  for (int rh = 0; rh < 2; ++rh) {
    aq[rh][0] = load_lhs_16x32(Q + (size_t)(t0 + rh * 16) * DH_ +  0, DH_);
    aq[rh][1] = load_lhs_16x32(Q + (size_t)(t0 + rh * 16) * DH_ + 32, DH_);
  }

  v8f o[2][4] = {};
  float m[2][8], l[2][8];
#pragma unroll
  for (int rh = 0; rh < 2; ++rh)
#pragma unroll
    for (int i = 0; i < 8; ++i) { m[rh][i] = -3.0e38f; l[rh][i] = 0.0f; }

  for (int j0 = 0; j0 < t0 + 32; j0 += 32) {     // causal: keys 0..t0+31
    // speculative prefetch of the next key/value block (global_prefetch_b8)
    __builtin_prefetch(K + (size_t)(j0 + 32) * DH_, 0, 1);
    __builtin_prefetch(VT + (size_t)(j0 + 32), 0, 1);

    // ---- scores: 2 row-halves x 2 col-halves, K tiles shared across halves
    v16h bk00 = load_lhs_16x32(K + (size_t)(j0     ) * DH_ +  0, DH_);
    v16h bk01 = load_lhs_16x32(K + (size_t)(j0     ) * DH_ + 32, DH_);
    v16h bk10 = load_lhs_16x32(K + (size_t)(j0 + 16) * DH_ +  0, DH_);
    v16h bk11 = load_lhs_16x32(K + (size_t)(j0 + 16) * DH_ + 32, DH_);
    v8f s[2][2] = {};
#pragma unroll
    for (int rh = 0; rh < 2; ++rh) {
      s[rh][0] = wmma16x16x32(aq[rh][0], bk00, s[rh][0]);
      s[rh][0] = wmma16x16x32(aq[rh][1], bk01, s[rh][0]);
      s[rh][1] = wmma16x16x32(aq[rh][0], bk10, s[rh][1]);
      s[rh][1] = wmma16x16x32(aq[rh][1], bk11, s[rh][1]);
    }

    // ---- causal mask + online softmax (per row-half) ----
#pragma unroll
    for (int rh = 0; rh < 2; ++rh) {
#pragma unroll
      for (int i = 0; i < 8; ++i) {
        const int row = t0 + rh * 16 + mb + i;
        if (j0 + n      > row) s[rh][0][i] = -3.0e38f;
        if (j0 + 16 + n > row) s[rh][1][i] = -3.0e38f;
      }
#pragma unroll
      for (int i = 0; i < 8; ++i) {
        float v = fmaxf(s[rh][0][i], s[rh][1][i]);
        v = fmaxf(v, __shfl_xor(v, 1, 32));
        v = fmaxf(v, __shfl_xor(v, 2, 32));
        v = fmaxf(v, __shfl_xor(v, 4, 32));
        v = fmaxf(v, __shfl_xor(v, 8, 32));
        const float mn    = fmaxf(m[rh][i], v);
        const float alpha = __expf(m[rh][i] - mn);
        const float p0 = __expf(s[rh][0][i] - mn);
        const float p1 = __expf(s[rh][1][i] - mn);
        float rs = p0 + p1;
        rs += __shfl_xor(rs, 1, 32);
        rs += __shfl_xor(rs, 2, 32);
        rs += __shfl_xor(rs, 4, 32);
        rs += __shfl_xor(rs, 8, 32);
        l[rh][i] = l[rh][i] * alpha + rs;
        m[rh][i] = mn;
        o[rh][0][i] *= alpha; o[rh][1][i] *= alpha;
        o[rh][2][i] *= alpha; o[rh][3][i] *= alpha;
        pbuf[(rh * 16 + mb + i) * 40 + n]      = (_Float16)p0;
        pbuf[(rh * 16 + mb + i) * 40 + 16 + n] = (_Float16)p1;
      }
    }

    // ---- O += P (32x32) * V (32x64); V^T tiles load as contiguous runs ----
    const v16h ap0 = load_lhs_16x32(pbuf,            40);  // rows 0..15
    const v16h ap1 = load_lhs_16x32(pbuf + 16 * 40,  40);  // rows 16..31
#pragma unroll
    for (int c = 0; c < 4; ++c) {
      // B[k][n] = V[j0+k][c*16+n] = VT[(c*16+n)][j0+k] -> A-pattern load
      const v16h bv = load_lhs_16x32(VT + (size_t)(c * 16) * T_ + j0, T_);
      o[0][c] = wmma16x16x32(ap0, bv, o[0][c]);
      o[1][c] = wmma16x16x32(ap1, bv, o[1][c]);
    }
  }

  // ---- normalize and store O as f16 in (B,T,D) with column offset h*64 ----
  const int b = bh >> 4, h = bh & 15;
#pragma unroll
  for (int rh = 0; rh < 2; ++rh) {
    _Float16* outp = oh + ((size_t)b * T_ + t0 + rh * 16) * D_ + h * DH_;
#pragma unroll
    for (int i = 0; i < 8; ++i) {
      const float inv = 1.0f / l[rh][i];
#pragma unroll
      for (int c = 0; c < 4; ++c)
        outp[(size_t)(mb + i) * D_ + c * 16 + n] = (_Float16)(o[rh][c][i] * inv);
    }
  }
}

// ---------------------------------------------------------------------------
// Kernel 3: output projection, fp32 store. Same 32x64-per-wave blocking.
// ---------------------------------------------------------------------------
__global__ __launch_bounds__(32)
void proj_out_kernel(const _Float16* __restrict__ oh, const _Float16* __restrict__ wo,
                     float* __restrict__ out) {
  const int row0 = blockIdx.x * 32;
  const int e0   = blockIdx.y * 64;

  v8f acc[2][4] = {};
  for (int k0 = 0; k0 < D_; k0 += 32) {
    const v16h a0 = load_lhs_16x32(oh + (size_t)row0 * D_ + k0, D_);
    const v16h a1 = load_lhs_16x32(oh + (size_t)(row0 + 16) * D_ + k0, D_);
#pragma unroll
    for (int c = 0; c < 4; ++c) {
      const v16h b = load_lhs_16x32(wo + (size_t)(e0 + c * 16) * D_ + k0, D_);
      acc[0][c] = wmma16x16x32(a0, b, acc[0][c]);
      acc[1][c] = wmma16x16x32(a1, b, acc[1][c]);
    }
  }

  const int lane = threadIdx.x & 31;
  const int n  = lane & 15;
  const int mb = (lane >> 4) << 3;
#pragma unroll
  for (int rh = 0; rh < 2; ++rh)
#pragma unroll
    for (int c = 0; c < 4; ++c)
#pragma unroll
      for (int i = 0; i < 8; ++i)
        out[(size_t)(row0 + rh * 16 + mb + i) * D_ + e0 + c * 16 + n] = acc[rh][c][i];
}

// ---------------------------------------------------------------------------
extern "C" void kernel_launch(void* const* d_in, const int* in_sizes, int n_in,
                              void* d_out, int out_size, void* d_ws, size_t ws_size,
                              hipStream_t stream) {
  const float* x  = (const float*)d_in[0];
  const float* wq = (const float*)d_in[1];
  const float* wk = (const float*)d_in[2];
  const float* wv = (const float*)d_in[3];
  const float* wo = (const float*)d_in[4];
  float* out = (float*)d_out;

  const size_t NTD = (size_t)B_ * T_ * D_;       // 4,194,304 elems
  const size_t NW  = (size_t)D_ * D_;            // 1,048,576 elems
  _Float16* xh  = (_Float16*)d_ws;               // (B,T,D)    f16
  _Float16* wqh = xh  + NTD;
  _Float16* wkh = wqh + NW;
  _Float16* wvh = wkh + NW;
  _Float16* woh = wvh + NW;
  _Float16* qh  = woh + NW;                      // (B,H,T,DH) f16, pre-scaled
  _Float16* kh  = qh  + NTD;                     // (B,H,T,DH) f16
  _Float16* vth = kh  + NTD;                     // (B,H,DH,T) f16  (V transposed)
  _Float16* oh  = vth + NTD;                     // (B,T,D)    f16   (48 MB total)

  // one-time fp32 -> f16 conversion
  cvt_f32_to_f16_kernel<<<dim3(NTD / 4 / 256), 256, 0, stream>>>(x,  xh,  (int)(NTD / 4));
  cvt_f32_to_f16_kernel<<<dim3(NW  / 4 / 256), 256, 0, stream>>>(wq, wqh, (int)(NW / 4));
  cvt_f32_to_f16_kernel<<<dim3(NW  / 4 / 256), 256, 0, stream>>>(wk, wkh, (int)(NW / 4));
  cvt_f32_to_f16_kernel<<<dim3(NW  / 4 / 256), 256, 0, stream>>>(wv, wvh, (int)(NW / 4));
  cvt_f32_to_f16_kernel<<<dim3(NW  / 4 / 256), 256, 0, stream>>>(wo, woh, (int)(NW / 4));

  const float qscale = 0.125f;                   // 1/sqrt(DH), folded into Q

  dim3 gridP(B_ * T_ / 32, H_);                  // (128, 16)
  proj_qkv_kernel<<<gridP, 32, 0, stream>>>(xh, wqh, qh, qscale);
  proj_qkv_kernel<<<gridP, 32, 0, stream>>>(xh, wkh, kh, 1.0f);
  proj_vT_kernel <<<gridP, 32, 0, stream>>>(xh, wvh, vth);

  dim3 gridF(T_ / 32, B_ * H_);                  // (64, 32)
  flash_attn_kernel<<<gridF, 32, 0, stream>>>(qh, kh, vth, oh);

  dim3 gridO(B_ * T_ / 32, D_ / 64);             // (128, 16)
  proj_out_kernel<<<gridO, 32, 0, stream>>>(oh, woh, out);
}
